// VectorQuantizer_18915035972113
// MI455X (gfx1250) — matrix operand order repaired and checked by reading, not compiled, not run on hardware
//
#include <hip/hip_runtime.h>
#include <hip/hip_bf16.h>

// ---------- types for WMMA ----------
typedef __attribute__((ext_vector_type(16))) __bf16 v16bf;
typedef __attribute__((ext_vector_type(8)))  float  v8f;
typedef __attribute__((ext_vector_type(8)))  unsigned int v8u;

union Frag {
    v8u          u;
    v16bf        b;
    uint4        u4[2];
    unsigned int w[8];
};

// ---------- bf16 helpers (round-to-nearest-even) ----------
static __device__ __forceinline__ unsigned short f32_to_bf16(float f) {
    unsigned int u = __float_as_uint(f);
    u += 0x7FFFu + ((u >> 16) & 1u);
    return (unsigned short)(u >> 16);
}
static __device__ __forceinline__ float bf16_to_f32(unsigned short h) {
    return __uint_as_float(((unsigned int)h) << 16);
}

// Problem constants (from reference): B=32, C=64, H=W=64, N_E=1024
#define HW      4096          // H*W
#define CDIM    64
#define NE      1024
#define NROWS   131072        // B*H*W
#define NELEM   8388608       // B*C*H*W
#define NJT     64            // NE/16
#define RTILES  4             // row-tiles (of 16 rows) per wave

// ============================================================
// 1) Pack codebook into WMMA B-fragment order, hi/lo bf16 split.
//    Index: ((jt*2 + kc)*32 + lane)*16 + t
//    B layout (32x16 bf16): K = (t<8 ? t : t+8) + (lane>=16 ? 8 : 0) (within kc chunk)
// ============================================================
__global__ void vq_prep_pack(const float* __restrict__ cb,
                             unsigned short* __restrict__ cb_hi,
                             unsigned short* __restrict__ cb_lo) {
    int u  = blockIdx.x * blockDim.x + threadIdx.x;   // 0..65535
    int t  = u & 15;
    int ln = (u >> 4) & 31;
    int kc = (u >> 9) & 1;
    int jt = u >> 10;
    int j  = jt * 16 + (ln & 15);
    int c  = kc * 32 + ((t < 8) ? t : t + 8) + ((ln & 16) ? 8 : 0);
    float x = cb[j * CDIM + c];
    unsigned short h = f32_to_bf16(x);
    cb_hi[u] = h;
    cb_lo[u] = f32_to_bf16(x - bf16_to_f32(h));
}

// ============================================================
// 2) Codebook squared norms
// ============================================================
__global__ void vq_prep_norm(const float* __restrict__ cb, float* __restrict__ cnorm) {
    int j = blockIdx.x * blockDim.x + threadIdx.x;
    if (j < NE) {
        float s = 0.f;
        #pragma unroll
        for (int c = 0; c < CDIM; ++c) {
            float v = cb[j * CDIM + c];
            s = fmaf(v, v, s);
        }
        cnorm[j] = s;
    }
}

// ============================================================
// 3) Main argmin kernel. 256 threads = 8 waves. Each wave owns
//    RTILES row-tiles of 16 rows; loops over 64 column tiles of
//    16 codes; per (row-tile, jt): 2 independent chains of 3
//    v_wmma_f32_16x16x32_bf16 (bf16x3 split-precision GEMM).
// ============================================================
__global__ void __launch_bounds__(256)
vq_argmin(const float* __restrict__ z,
          const unsigned short* __restrict__ cb_hi,
          const unsigned short* __restrict__ cb_lo,
          const float* __restrict__ cnorm,
          int* __restrict__ idxOut) {
    const int tid    = threadIdx.x;
    const int ln     = tid & 31;
    const int wave   = tid >> 5;
    const int lm     = ln & 15;
    const int hiHalf = (ln >> 4) & 1;      // 0: rows 0..7, 1: rows 8..15 of C/D

    // ---- build A fragments (hi/lo bf16) for RTILES row-tiles ----
    Frag aHi[RTILES][2], aLo[RTILES][2];   // [row-tile][kc]
    #pragma unroll
    for (int rr = 0; rr < RTILES; ++rr) {
        const int rt = blockIdx.x * (8 * RTILES) + wave * RTILES + rr;  // row-tile id
        const int nb = rt * 16;                           // first row (flat n)
        const int b  = nb >> 12;                          // batch
        const int s  = (nb & (HW - 1)) + lm;              // spatial pos of my row
        const float* zb = z + (size_t)b * (CDIM * HW) + s;
        #pragma unroll
        for (int kc = 0; kc < 2; ++kc) {
            unsigned short th[16], tl[16];
            #pragma unroll
            for (int t = 0; t < 16; ++t) {
                // A layout (16x32 bf16): K = (t<8 ? t : t+8) + (hiHalf?8:0)
                int c = kc * 32 + ((t < 8) ? t : t + 8) + (hiHalf ? 8 : 0);
                float x = zb[c * HW];
                th[t] = f32_to_bf16(x);
                tl[t] = f32_to_bf16(x - bf16_to_f32(th[t]));
            }
            #pragma unroll
            for (int v = 0; v < 8; ++v) {
                aHi[rr][kc].w[v] = (unsigned int)th[2 * v] | ((unsigned int)th[2 * v + 1] << 16);
                aLo[rr][kc].w[v] = (unsigned int)tl[2 * v] | ((unsigned int)tl[2 * v + 1] << 16);
            }
        }
    }

    float bestd[RTILES][8];
    int   bestj[RTILES][8];
    #pragma unroll
    for (int rr = 0; rr < RTILES; ++rr)
        #pragma unroll
        for (int r = 0; r < 8; ++r) { bestd[rr][r] = 3.4e38f; bestj[rr][r] = 0; }

    const uint4* bhp = (const uint4*)cb_hi;
    const uint4* blp = (const uint4*)cb_lo;

    for (int jt = 0; jt < NJT; ++jt) {
        // ---- load B fragments (pre-swizzled): 32B contiguous per lane ----
        Frag bHi[2], bLo[2];
        #pragma unroll
        for (int kc = 0; kc < 2; ++kc) {
            int fo = (((jt * 2 + kc) * 32) + ln) * 2;     // uint4 units
            bHi[kc].u4[0] = bhp[fo];     bHi[kc].u4[1] = bhp[fo + 1];
            bLo[kc].u4[0] = blp[fo];     bLo[kc].u4[1] = blp[fo + 1];
        }
        const float cn = cnorm[jt * 16 + lm];
        const int   j  = jt * 16 + lm;

        #pragma unroll
        for (int rr = 0; rr < RTILES; ++rr) {
            // Two independent accumulation chains (one per K-chunk) so the
            // matrix pipe can overlap WMMAs within and across row-tiles.
            v8f acc0 = {0.f, 0.f, 0.f, 0.f, 0.f, 0.f, 0.f, 0.f};
            v8f acc1 = {0.f, 0.f, 0.f, 0.f, 0.f, 0.f, 0.f, 0.f};
            // bf16x3: hi*hi + hi*lo + lo*hi  (f32 accumulate)
            acc0 = __builtin_amdgcn_wmma_f32_16x16x32_bf16(
                       false, aHi[rr][0].b, false, bHi[0].b, (short)0, acc0, false, false);
            acc1 = __builtin_amdgcn_wmma_f32_16x16x32_bf16(
                       false, aHi[rr][1].b, false, bHi[1].b, (short)0, acc1, false, false);
            acc0 = __builtin_amdgcn_wmma_f32_16x16x32_bf16(
                       false, aHi[rr][0].b, false, bLo[0].b, (short)0, acc0, false, false);
            acc1 = __builtin_amdgcn_wmma_f32_16x16x32_bf16(
                       false, aHi[rr][1].b, false, bLo[1].b, (short)0, acc1, false, false);
            acc0 = __builtin_amdgcn_wmma_f32_16x16x32_bf16(
                       false, aLo[rr][0].b, false, bHi[0].b, (short)0, acc0, false, false);
            acc1 = __builtin_amdgcn_wmma_f32_16x16x32_bf16(
                       false, aLo[rr][1].b, false, bHi[1].b, (short)0, acc1, false, false);
            // dist (minus const ||z||^2) = ||e_j||^2 - 2 * dot
            #pragma unroll
            for (int r = 0; r < 8; ++r) {
                float d = fmaf(-2.0f, acc0[r] + acc1[r], cn);
                if (d < bestd[rr][r]) { bestd[rr][r] = d; bestj[rr][r] = j; }  // jt ascending -> first index wins ties
            }
        }
    }

    // ---- butterfly argmin across the 16 lanes that share each row group ----
    #pragma unroll
    for (int rr = 0; rr < RTILES; ++rr) {
        #pragma unroll
        for (int m = 1; m < 16; m <<= 1) {
            #pragma unroll
            for (int r = 0; r < 8; ++r) {
                float od = __shfl_xor(bestd[rr][r], m, 32);
                int   oj = __shfl_xor(bestj[rr][r], m, 32);
                if (od < bestd[rr][r] || (od == bestd[rr][r] && oj < bestj[rr][r])) {
                    bestd[rr][r] = od; bestj[rr][r] = oj;
                }
            }
        }
        if (lm == 0) {
            const int rt   = blockIdx.x * (8 * RTILES) + wave * RTILES + rr;
            const int base = rt * 16 + (hiHalf ? 8 : 0);   // C/D: lane>=16 holds M=8..15
            #pragma unroll
            for (int r = 0; r < 8; ++r) idxOut[base + r] = bestj[rr][r];
        }
    }
}

// ============================================================
// 4) Gather + straight-through output + deterministic partial sums
// ============================================================
__global__ void __launch_bounds__(256)
vq_gather(const float* __restrict__ z, const float* __restrict__ cb,
          const int* __restrict__ idx, float* __restrict__ out,
          float* __restrict__ partials) {
    __shared__ float sdata[256];
    const int tid = threadIdx.x;
    const int i   = blockIdx.x * 256 + tid;       // NCHW flat index
    const int s   = i & (HW - 1);
    const int c   = (i >> 12) & (CDIM - 1);
    const int b   = i >> 18;
    const int n   = (b << 12) + s;                // flat NHWC row
    const float zv   = z[i];
    const float zq   = cb[idx[n] * CDIM + c];
    const float diff = zq - zv;                   // matches reference op order
    out[1 + i] = zv + diff;                       // z + stop_grad(z_q - z)
    sdata[tid] = diff * diff;
    __syncthreads();
    #pragma unroll
    for (int sft = 128; sft > 0; sft >>= 1) {
        if (tid < sft) sdata[tid] += sdata[tid + sft];
        __syncthreads();
    }
    if (tid == 0) partials[blockIdx.x] = sdata[0];
}

// ============================================================
// 5) Final deterministic loss reduction: (1 + BETA) * mean
// ============================================================
__global__ void __launch_bounds__(256)
vq_loss(const float* __restrict__ partials, float* __restrict__ out) {
    __shared__ float sdata[256];
    const int tid = threadIdx.x;
    float s = 0.f;
    for (int i = tid; i < (NELEM / 256); i += 256) s += partials[i];  // fixed order
    sdata[tid] = s;
    __syncthreads();
    #pragma unroll
    for (int sft = 128; sft > 0; sft >>= 1) {
        if (tid < sft) sdata[tid] += sdata[tid + sft];
        __syncthreads();
    }
    if (tid == 0) out[0] = 1.25f * sdata[0] * (1.0f / (float)NELEM);
}

// ============================================================
extern "C" void kernel_launch(void* const* d_in, const int* in_sizes, int n_in,
                              void* d_out, int out_size, void* d_ws, size_t ws_size,
                              hipStream_t stream) {
    const float* z  = (const float*)d_in[0];   // [32,64,64,64] f32
    const float* cb = (const float*)d_in[1];   // [1024,64] f32
    float* out = (float*)d_out;                // [1 + 8388608] f32

    char* ws = (char*)d_ws;
    unsigned short* cb_hi    = (unsigned short*)(ws);              // 128 KB
    unsigned short* cb_lo    = (unsigned short*)(ws + 131072);     // 128 KB
    float*          cnorm    = (float*)(ws + 262144);              // 4 KB
    int*            idx      = (int*)(ws + 266240);                // 512 KB
    float*          partials = (float*)(ws + 790528);              // 128 KB

    vq_prep_pack<<<256, 256, 0, stream>>>(cb, cb_hi, cb_lo);
    vq_prep_norm<<<4, 256, 0, stream>>>(cb, cnorm);
    vq_argmin<<<NROWS / (16 * 8 * RTILES), 256, 0, stream>>>(z, cb_hi, cb_lo, cnorm, idx);
    vq_gather<<<NELEM / 256, 256, 0, stream>>>(z, cb, idx, out, partials);
    vq_loss<<<1, 256, 0, stream>>>(partials, out);
}
